// Attention_19602230739759
// MI455X (gfx1250) — matrix operand order repaired and checked by reading, not compiled, not run on hardware
//
#include <hip/hip_runtime.h>
#include <hip/hip_bf16.h>

typedef __bf16 bf16;
typedef __attribute__((ext_vector_type(16))) __bf16 v16bf;
typedef __attribute__((ext_vector_type(8)))  __bf16 v8bf;
typedef __attribute__((ext_vector_type(4)))  __bf16 v4bf;
typedef __attribute__((ext_vector_type(8)))  float  v8f;

#define DIM    512
#define HEADS  16
#define HD     32
#define N_TOK  64
#define NW     256
#define SCALE  0.17677669529663687f   // 32^-0.5

// LDS strides (elements); all give 8-byte-aligned rows + conflict-free 16-lane access
#define XS_LD  516
#define QK_LD  36
#define VT_LD  68
#define PS_LD  68

#define XS_ELEMS (N_TOK * XS_LD)           // 33024
#define QK_ELEMS (HEADS * N_TOK * QK_LD)   // 36864
#define VT_ELEMS (HEADS * HD * VT_LD)      // 34816
#define PS_ELEMS (HEADS * 16 * PS_LD)      // 17408
#define SMEM_BYTES ((XS_ELEMS + 2 * QK_ELEMS + VT_ELEMS + PS_ELEMS) * 2)  // 317952 <= 320KB

static __device__ __forceinline__ bf16 f2bf(float f) {
  unsigned u = __builtin_bit_cast(unsigned, f);
  u += 0x7fffu + ((u >> 16) & 1u);           // round-to-nearest-even
  unsigned short h = (unsigned short)(u >> 16);
  return __builtin_bit_cast(bf16, h);
}

union V16U { v16bf v; v8bf o[2]; v4bf q[4]; };

// A operand 16x32 (MxK), row-major [row][k], stride ld.
// Contiguous k order (same as B operand): lanes 0-15 row m K=0..15, lanes 16-31 row m K=16..31.
static __device__ __forceinline__ v16bf load_a(const bf16* base, int ld, int m0, int k0, int lane) {
  int hl = (lane >> 4) & 1;
  int mr = lane & 15;
  const bf16* p = base + (m0 + mr) * ld + k0 + hl * 16;
  V16U A;
  A.q[0] = *(const v4bf*)(p);
  A.q[1] = *(const v4bf*)(p + 4);
  A.q[2] = *(const v4bf*)(p + 8);
  A.q[3] = *(const v4bf*)(p + 12);
  return A.v;
}

// B operand 32x16 (KxN) from N-major storage [n][k], stride ld.
// Contiguous k order (same as load_a): lanes 0-15 col n K=0..15, lanes 16-31 col n K=16..31. 8-byte loads (LDS).
static __device__ __forceinline__ v16bf load_b8(const bf16* base, int ld, int n0, int k0, int lane) {
  int hl = (lane >> 4) & 1;
  int n  = lane & 15;
  const bf16* p = base + (n0 + n) * ld + k0 + hl * 16;
  V16U B;
  B.q[0] = *(const v4bf*)(p);
  B.q[1] = *(const v4bf*)(p + 4);
  B.q[2] = *(const v4bf*)(p + 8);
  B.q[3] = *(const v4bf*)(p + 12);
  return B.v;
}

// Same, 16-byte loads for fully-aligned global (L2-resident) weights.
static __device__ __forceinline__ v16bf load_b16(const bf16* base, int ld, int n0, int k0, int lane) {
  int hl = (lane >> 4) & 1;
  int n  = lane & 15;
  const bf16* p = base + (n0 + n) * ld + k0 + hl * 16;
  V16U B;
  B.o[0] = *(const v8bf*)(p);
  B.o[1] = *(const v8bf*)(p + 8);
  return B.v;
}

// ---- weight prep: fp32 -> bf16, transposed to [out][in] ----
__global__ void prep_qkv_w(const float* __restrict__ Wq, const float* __restrict__ Wkv,
                           bf16* __restrict__ WqkvT) {
  int tid = blockIdx.x * 256 + threadIdx.x;     // 1536*512 total
  if (tid >= 1536 * 512) return;
  int o = tid >> 9, i = tid & 511;
  float v = (o < DIM) ? Wq[i * DIM + o] : Wkv[i * (2 * DIM) + (o - DIM)];
  WqkvT[o * DIM + i] = f2bf(v);
}

__global__ void prep_p_w(const float* __restrict__ Wp, bf16* __restrict__ WpT) {
  int tid = blockIdx.x * 256 + threadIdx.x;     // 512*512 total
  if (tid >= DIM * DIM) return;
  int o = tid >> 9, i = tid & 511;
  WpT[o * DIM + i] = f2bf(Wp[i * DIM + o]);
}

// ---- fused windowed attention: one workgroup per window ----
__global__ __launch_bounds__(512, 1)
void attn_fused(const float* __restrict__ x, const float* __restrict__ mask,
                const float* __restrict__ bq, const float* __restrict__ bkv,
                const float* __restrict__ bp,
                const bf16* __restrict__ WqkvT, const bf16* __restrict__ WpT,
                float* __restrict__ out)
{
  extern __shared__ bf16 smem[];
  bf16* xs = smem;                 // 64 x 512 staged x; reused as attn-out (os)
  bf16* qs = xs + XS_ELEMS;        // [16][64][32] q * SCALE
  bf16* ks = qs + QK_ELEMS;        // [16][64][32] k
  bf16* vT = ks + QK_ELEMS;        // [16][32][64] v transposed
  bf16* ps = vT + VT_ELEMS;        // [16 waves][16][64] softmax probs

  const int tid  = threadIdx.x;
  const int wave = tid >> 5;
  const int lane = tid & 31;
  const int hl   = (lane >> 4) & 1;
  const int nn   = lane & 15;
  const int b    = blockIdx.x;

  // ---- phase 1: stage x into LDS as bf16 ----
  {
    const float4* xv = (const float4*)(x + (size_t)b * (N_TOK * DIM));
    #pragma unroll
    for (int it = 0; it < 16; ++it) {
      int idx = it * 512 + tid;
      float4 f = xv[idx];
      int row = idx >> 7;
      int col = (idx & 127) << 2;
      bf16* d = xs + row * XS_LD + col;
      d[0] = f2bf(f.x); d[1] = f2bf(f.y); d[2] = f2bf(f.z); d[3] = f2bf(f.w);
    }
  }
  __syncthreads();

  // ---- phase 2: QKV projection, 64 x 1536, scatter to per-head LDS ----
  for (int t = wave; t < 384; t += 16) {          // 24 tiles per wave
    int m0 = (t & 3) << 4;
    int c0 = (t >> 2) << 4;
    v8f acc = {};
    #pragma unroll
    for (int kk = 0; kk < 16; ++kk) {
      v16bf a  = load_a(xs, XS_LD, m0, kk * 32, lane);
      v16bf bm = load_b16(WqkvT, DIM, c0, kk * 32, lane);
      acc = __builtin_amdgcn_wmma_f32_16x16x32_bf16(false, a, false, bm, (short)0, acc, false, false);
    }
    int c = c0 + nn;
    int d0 = c0 & 31;
    if (c0 < DIM) {                                // -> q (fold SCALE)
      float bias = bq[c];
      bf16* qh = qs + (c0 >> 5) * (N_TOK * QK_LD);
      #pragma unroll
      for (int r = 0; r < 8; ++r)
        qh[(m0 + r + 8 * hl) * QK_LD + d0 + nn] = f2bf((acc[r] + bias) * SCALE);
    } else if (c0 < 2 * DIM) {                     // -> k
      float bias = bkv[c - DIM];
      bf16* kh = ks + ((c0 - DIM) >> 5) * (N_TOK * QK_LD);
      #pragma unroll
      for (int r = 0; r < 8; ++r)
        kh[(m0 + r + 8 * hl) * QK_LD + d0 + nn] = f2bf(acc[r] + bias);
    } else {                                       // -> v, stored transposed [d][token]
      float bias = bkv[c - DIM];
      bf16* vh = vT + ((c0 - 2 * DIM) >> 5) * (HD * VT_LD);
      #pragma unroll
      for (int r = 0; r < 8; ++r)
        vh[(d0 + nn) * VT_LD + (m0 + r + 8 * hl)] = f2bf(acc[r] + bias);
    }
  }
  __syncthreads();

  // ---- phase 3: attention, one head per wave; write attn-out into xs ----
  {
    const int h = wave;
    const bf16* qh = qs + h * (N_TOK * QK_LD);
    const bf16* kh = ks + h * (N_TOK * QK_LD);
    const bf16* vh = vT + h * (HD * VT_LD);
    bf16* pw = ps + wave * (16 * PS_LD);
    const float* mw = mask + (size_t)(b & (NW - 1)) * (N_TOK * N_TOK);

    #pragma unroll
    for (int mt = 0; mt < 4; ++mt) {
      int m0 = mt << 4;
      float accS[4][8];
      v16bf aq = load_a(qh, QK_LD, m0, 0, lane);
      #pragma unroll
      for (int nt = 0; nt < 4; ++nt) {
        v8f s = {};
        v16bf bk = load_b8(kh, QK_LD, nt * 16, 0, lane);   // K=d=32: one WMMA
        s = __builtin_amdgcn_wmma_f32_16x16x32_bf16(false, aq, false, bk, (short)0, s, false, false);
        #pragma unroll
        for (int r = 0; r < 8; ++r)
          accS[nt][r] = s[r] + mw[(m0 + r + 8 * hl) * N_TOK + nt * 16 + nn];
      }
      // softmax over 64 cols: in-lane over 4 tiles + shfl_xor over 16-lane group
      float rs[8];
      #pragma unroll
      for (int r = 0; r < 8; ++r) {
        float m = fmaxf(fmaxf(accS[0][r], accS[1][r]), fmaxf(accS[2][r], accS[3][r]));
        #pragma unroll
        for (int off = 1; off < 16; off <<= 1)
          m = fmaxf(m, __shfl_xor(m, off, 32));
        float ssum = 0.f;
        #pragma unroll
        for (int nt = 0; nt < 4; ++nt) {
          float p = __expf(accS[nt][r] - m);
          accS[nt][r] = p;
          ssum += p;
        }
        #pragma unroll
        for (int off = 1; off < 16; off <<= 1)
          ssum += __shfl_xor(ssum, off, 32);
        rs[r] = ssum;
      }
      // stash P (bf16) in wave-private LDS strip
      #pragma unroll
      for (int nt = 0; nt < 4; ++nt)
        #pragma unroll
        for (int r = 0; r < 8; ++r)
          pw[(r + 8 * hl) * PS_LD + nt * 16 + nn] = f2bf(accS[nt][r]);
      // O = P @ V (K=64 tokens), normalize, write to os (= xs)
      #pragma unroll
      for (int nto = 0; nto < 2; ++nto) {
        v8f o = {};
        #pragma unroll
        for (int kt = 0; kt < 2; ++kt) {
          v16bf ap = load_a(pw, PS_LD, 0, kt * 32, lane);
          v16bf bv = load_b8(vh, VT_LD, nto * 16, kt * 32, lane);
          o = __builtin_amdgcn_wmma_f32_16x16x32_bf16(false, ap, false, bv, (short)0, o, false, false);
        }
        #pragma unroll
        for (int r = 0; r < 8; ++r) {
          float val = o[r] * __builtin_amdgcn_rcpf(rs[r]);
          xs[(m0 + r + 8 * hl) * XS_LD + h * HD + nto * 16 + nn] = f2bf(val);
        }
      }
    }
  }
  __syncthreads();

  // ---- phase 4: output projection, 64 x 512, fp32 store ----
  for (int t = wave; t < 128; t += 16) {           // 8 tiles per wave
    int m0 = (t & 3) << 4;
    int c0 = (t >> 2) << 4;
    v8f acc = {};
    #pragma unroll
    for (int kk = 0; kk < 16; ++kk) {
      v16bf a  = load_a(xs, XS_LD, m0, kk * 32, lane);
      v16bf bw = load_b16(WpT, DIM, c0, kk * 32, lane);
      acc = __builtin_amdgcn_wmma_f32_16x16x32_bf16(false, a, false, bw, (short)0, acc, false, false);
    }
    float bias = bp[c0 + nn];
    #pragma unroll
    for (int r = 0; r < 8; ++r) {
      size_t off = ((size_t)b * N_TOK + m0 + r + 8 * hl) * DIM + c0 + nn;
      out[off] = acc[r] + bias;
    }
  }
}

extern "C" void kernel_launch(void* const* d_in, const int* in_sizes, int n_in,
                              void* d_out, int out_size, void* d_ws, size_t ws_size,
                              hipStream_t stream) {
  (void)in_sizes; (void)n_in; (void)out_size; (void)ws_size;
  const float* x    = (const float*)d_in[0];
  const float* mask = (const float*)d_in[1];
  const float* Wq   = (const float*)d_in[2];
  const float* bq   = (const float*)d_in[3];
  const float* Wkv  = (const float*)d_in[4];
  const float* bkv  = (const float*)d_in[5];
  const float* Wp   = (const float*)d_in[6];
  const float* bp   = (const float*)d_in[7];
  float* out = (float*)d_out;

  bf16* WqkvT = (bf16*)d_ws;                   // 1536*512 bf16 = 1.5 MB
  bf16* WpT   = WqkvT + 1536 * 512;            // 512*512 bf16  = 0.5 MB

  (void)hipFuncSetAttribute((const void*)attn_fused,
                            hipFuncAttributeMaxDynamicSharedMemorySize, SMEM_BYTES);

  prep_qkv_w<<<(1536 * 512 + 255) / 256, 256, 0, stream>>>(Wq, Wkv, WqkvT);
  prep_p_w<<<(DIM * DIM + 255) / 256, 256, 0, stream>>>(Wp, WpT);
  attn_fused<<<1024, 512, SMEM_BYTES, stream>>>(x, mask, bq, bkv, bp, WqkvT, WpT, out);
}
